// SpanV2_45045617000956
// MI455X (gfx1250) — compile-verified, matrix operand order checked by reading, not run on hardware
//
#include <hip/hip_runtime.h>

typedef __attribute__((ext_vector_type(16))) __bf16 v16bf;
typedef __attribute__((ext_vector_type(8)))  __bf16 v8bf;
typedef __attribute__((ext_vector_type(4)))  __bf16 v4bf;
typedef __attribute__((ext_vector_type(8)))  float  v8f;

#define TM     32      // spans per workgroup
#define IN_DIM 1686    // 2*768 + 150
#define KP     1696    // padded K (53 * 32)
#define KC1    53      // K-chunks of 32 for GEMM1
#define NT1    48      // 768 / 16 N-tiles for GEMM1
#define NTW    6       // N-tiles per wave (x 2 M-halves)
#define HDIM   768
#define HP     784     // padded h row stride (bf16 elems)
#define KC2    24      // 768 / 32 K-chunks for GEMM2
#define NLOG   25
#define SEQ    512
#define WEMB_D 150

__device__ __forceinline__ __bf16 f2bf(float f) { return (__bf16)f; }

// ---- Prep: W1 (1686x768 f32) -> bf16 B-fragments [KC1][NT1][32 lanes][16 elems]
extern "C" __global__ __launch_bounds__(256)
void prep_w1(const float* __restrict__ W1, __bf16* __restrict__ w1f) {
    const int idx = blockIdx.x * 256 + threadIdx.x;
    const int total = KC1 * NT1 * 32 * 16;
    if (idx >= total) return;
    const int e    = idx & 15;
    const int lane = (idx >> 4) & 31;
    const int frag = idx >> 9;
    const int nt   = frag % NT1;
    const int kc   = frag / NT1;
    // B 32x16 bf16 layout: lanes 0-15 -> K 0-7 & 16-23 of chunk; lanes 16-31 -> K 8-15 & 24-31
    const int k = kc * 32 + ((lane >> 4) << 3) + (e < 8 ? e : e + 8);
    const int n = nt * 16 + (lane & 15);
    const float v = (k < IN_DIM) ? W1[(size_t)k * HDIM + n] : 0.0f;
    w1f[idx] = f2bf(v);
}

// ---- Prep: W2 (768x25 f32) -> bf16 B-fragments [KC2][2][32][16], N padded to 32
extern "C" __global__ __launch_bounds__(256)
void prep_w2(const float* __restrict__ W2, __bf16* __restrict__ w2f) {
    const int idx = blockIdx.x * 256 + threadIdx.x;
    const int total = KC2 * 2 * 32 * 16;
    if (idx >= total) return;
    const int e    = idx & 15;
    const int lane = (idx >> 4) & 31;
    const int frag = idx >> 9;
    const int nt   = frag & 1;
    const int kc   = frag >> 1;
    const int k = kc * 32 + ((lane >> 4) << 3) + (e < 8 ? e : e + 8);
    const int n = nt * 16 + (lane & 15);
    const float v = (n < NLOG) ? W2[(size_t)k * NLOG + n] : 0.0f;
    w2f[idx] = f2bf(v);
}

// ---- Fused: gather -> GEMM1(bf16 WMMA) + bias + ReLU -> GEMM2(bf16 WMMA) + bias
extern "C" __global__ __launch_bounds__(256)
void span_fused_kernel(const float* __restrict__ hidden,
                       const long long* __restrict__ spans,
                       const float* __restrict__ wemb,
                       const float* __restrict__ b1,
                       const float* __restrict__ b2,
                       const __bf16* __restrict__ w1f,
                       const __bf16* __restrict__ w2f,
                       float* __restrict__ out)
{
    // LDS: A-tile (108.5 KB); h-tile (50 KB) + reduce (16 KB) alias it after GEMM1
    __shared__ __align__(16) char smem[TM * KP * 2];
    __shared__ int sIdx[TM][3];
    __bf16 (*As)[KP] = (__bf16 (*)[KP])smem;
    __bf16 (*Hs)[HP] = (__bf16 (*)[HP])smem;
    float (*Red)[4][16][32] = (float (*)[4][16][32])(smem + TM * HP * 2);

    const int tid  = threadIdx.x;
    const int lane = tid & 31;
    const int wave = tid >> 5;
    const int spanBase = blockIdx.x * TM;        // 4096 % TM == 0 -> no batch straddle
    const int bIdx = spanBase >> 12;

    if (tid < TM) {
        const long long* sp = spans + (size_t)(spanBase + tid) * 3;
        sIdx[tid][0] = (int)sp[0];
        sIdx[tid][1] = (int)sp[1];
        sIdx[tid][2] = (int)sp[2];
    }
    __syncthreads();

    // Gather span embeddings -> bf16 A-tile in LDS. float4 loads + packed 4xbf16 stores;
    // segment boundaries (768, 1536) are 16B-aligned, width-emb tail uses scalar guards.
    const float* hsB = hidden + (size_t)bIdx * SEQ * HDIM;
    for (int m = 0; m < TM; ++m) {
        const float* rs = hsB + sIdx[m][0] * HDIM;
        const float* re = hsB + sIdx[m][1] * HDIM;
        const float* rw = wemb + sIdx[m][2] * WEMB_D;
        for (int c = tid * 4; c < KP; c += 1024) {
            float4 v;
            if (c < 768)       v = *(const float4*)(rs + c);
            else if (c < 1536) v = *(const float4*)(re + c - 768);
            else {
                const int o = c - 1536;
                v.x = (o + 0 < WEMB_D) ? rw[o + 0] : 0.0f;
                v.y = (o + 1 < WEMB_D) ? rw[o + 1] : 0.0f;
                v.z = (o + 2 < WEMB_D) ? rw[o + 2] : 0.0f;
                v.w = (o + 3 < WEMB_D) ? rw[o + 3] : 0.0f;
            }
            v4bf p = { f2bf(v.x), f2bf(v.y), f2bf(v.z), f2bf(v.w) };
            *(v4bf*)&As[m][c] = p;
        }
    }
    __syncthreads();

    const int row = lane & 15;          // M (or N) within 16
    const int lo  = (lane >> 4) << 3;   // 0 or 8 (lane-half K offset)
    const int ng  = wave * NTW;         // first N-tile of this wave

    v8f z;
    #pragma unroll
    for (int i = 0; i < 8; ++i) z[i] = 0.0f;

    // ---- GEMM1: each wave owns 6 N-tiles x BOTH 16-row M-halves.
    // Every W1 fragment is fetched once per wave and reused for 2 WMMAs ->
    // per-WG B traffic equals the unique 2.6 MB (no duplicate L2 fetches).
    v8f acc[2][NTW];
    #pragma unroll
    for (int t = 0; t < NTW; ++t) { acc[0][t] = z; acc[1][t] = z; }

    const __bf16* arow0 = &As[row][0];
    const __bf16* arow1 = &As[16 + row][0];
    for (int kc = 0; kc < KC1; ++kc) {
        const int k0 = kc * 32 + lo;
        union { v16bf v; v8bf h[2]; } a0, a1;
        a0.h[0] = *(const v8bf*)(arow0 + k0);        // K lo..lo+7
        a0.h[1] = *(const v8bf*)(arow0 + k0 + 16);   // K lo+16..lo+23
        a1.h[0] = *(const v8bf*)(arow1 + k0);
        a1.h[1] = *(const v8bf*)(arow1 + k0 + 16);
        const v16bf* bp = ((const v16bf*)w1f) + (size_t)(kc * NT1 + ng) * 32 + lane;
        #pragma unroll
        for (int t = 0; t < NTW; ++t) {
            const v16bf bfrag = bp[t * 32];
            acc[0][t] = __builtin_amdgcn_wmma_f32_16x16x32_bf16(
                false, a0.v, false, bfrag, (short)0, acc[0][t], false, false);
            acc[1][t] = __builtin_amdgcn_wmma_f32_16x16x32_bf16(
                false, a1.v, false, bfrag, (short)0, acc[1][t], false, false);
        }
    }
    __syncthreads();   // all waves done reading As before Hs aliases it

    // bias + ReLU, write h tile to LDS as bf16. C layout: M = r + lo, N = row.
    #pragma unroll
    for (int t = 0; t < NTW; ++t) {
        const int col = (ng + t) * 16 + row;
        const float bv = b1[col];
        #pragma unroll
        for (int mh = 0; mh < 2; ++mh) {
            #pragma unroll
            for (int r = 0; r < 8; ++r) {
                float v = acc[mh][t][r] + bv;
                v = v > 0.0f ? v : 0.0f;
                Hs[mh * 16 + r + lo][col] = f2bf(v);
            }
        }
    }
    __syncthreads();

    // ---- GEMM2: logits = h @ W2 + b2 (K split across 4 waves per M-half)
    const int mhalf = wave >> 2;
    v8f acc2[2];
    acc2[0] = z; acc2[1] = z;
    const __bf16* hrow = &Hs[mhalf * 16 + row][0];
    #pragma unroll
    for (int j = 0; j < 6; ++j) {
        const int kc = (wave & 3) * 6 + j;
        const int k0 = kc * 32 + lo;
        union { v16bf v; v8bf h[2]; } a;
        a.h[0] = *(const v8bf*)(hrow + k0);
        a.h[1] = *(const v8bf*)(hrow + k0 + 16);
        const v16bf* bp = ((const v16bf*)w2f) + (size_t)(kc * 2) * 32 + lane;
        acc2[0] = __builtin_amdgcn_wmma_f32_16x16x32_bf16(
            false, a.v, false, bp[0],  (short)0, acc2[0], false, false);
        acc2[1] = __builtin_amdgcn_wmma_f32_16x16x32_bf16(
            false, a.v, false, bp[32], (short)0, acc2[1], false, false);
    }
    #pragma unroll
    for (int nt = 0; nt < 2; ++nt)
        #pragma unroll
        for (int r = 0; r < 8; ++r)
            Red[mhalf][wave & 3][r + lo][nt * 16 + row] = acc2[nt][r];
    __syncthreads();

    // Reduce 4 K-partials, add b2, write fp32 logits
    for (int i = tid; i < TM * 32; i += 256) {
        const int m = i >> 5;
        const int l = i & 31;
        if (l < NLOG) {
            float s = b2[l];
            #pragma unroll
            for (int w = 0; w < 4; ++w) s += Red[m >> 4][w][m & 15][l];
            out[(size_t)(spanBase + m) * NLOG + l] = s;
        }
    }
}

extern "C" void kernel_launch(void* const* d_in, const int* in_sizes, int n_in,
                              void* d_out, int out_size, void* d_ws, size_t ws_size,
                              hipStream_t stream) {
    const float*     hidden = (const float*)d_in[0];
    const long long* spans  = (const long long*)d_in[1];
    const float*     wemb   = (const float*)d_in[2];
    const float*     W1     = (const float*)d_in[3];
    const float*     b1     = (const float*)d_in[4];
    const float*     W2     = (const float*)d_in[5];
    const float*     b2     = (const float*)d_in[6];
    float* out = (float*)d_out;

    const size_t w1fElems = (size_t)KC1 * NT1 * 32 * 16;   // 1,302,528 bf16 (~2.6 MB)
    const size_t w2fElems = (size_t)KC2 * 2 * 32 * 16;     // 24,576 bf16
    __bf16* w1f = (__bf16*)d_ws;
    __bf16* w2f = (__bf16*)((char*)d_ws + w1fElems * sizeof(__bf16));

    prep_w1<<<(int)((w1fElems + 255) / 256), 256, 0, stream>>>(W1, w1f);
    prep_w2<<<(int)((w2fElems + 255) / 256), 256, 0, stream>>>(W2, w2f);

    // 65536 spans / 32 per WG = 2048 workgroups, 8 waves each
    span_fused_kernel<<<2048, 256, 0, stream>>>(hidden, spans, wemb, b1, b2, w1f, w2f, out);
}
